// CoAttention_71511205479003
// MI455X (gfx1250) — matrix operand order repaired and checked by reading, not compiled, not run on hardware
//
#include <hip/hip_runtime.h>
#include <hip/hip_bf16.h>
#include <math.h>

// ---------------------------------------------------------------------------
// CDNA5 (gfx1250) CoAttention:
//   - all GEMMs on v_wmma_f32_16x16x32_bf16 (wave32)
//   - B tiles staged through LDS with GLOBAL_LOAD_ASYNC_TO_LDS_B128
//     (ASYNCcnt double-buffering), read back with ds_load_b128
//   - compile-time K with peeled final step: branch-free steady-state loop
//   - all B sources row-major N x K (contiguous fragment loads); transposed
//     copies are materialized in epilogues/prep so no strided loads remain
// ---------------------------------------------------------------------------

typedef __attribute__((ext_vector_type(16))) __bf16 v16bf;
typedef __attribute__((ext_vector_type(8)))  __bf16 v8bf;
typedef __attribute__((ext_vector_type(8)))  float  v8f;
typedef int v4i_vs __attribute__((vector_size(16)));

#define BATCH 64
#define TT    384      // T = H/2
#define HH    768
#define OC    2304     // output row width = 1536 + 768

#if __has_builtin(__builtin_amdgcn_global_load_async_to_lds_b128)
#define HAVE_ASYNC_LDS 1
#else
#define HAVE_ASYNC_LDS 0
#endif

// 16-byte global -> LDS copy (async when gfx1250 path is available)
__device__ inline void copy16_g2l(__bf16* lds_dst, const __bf16* gsrc) {
#if HAVE_ASYNC_LDS
  __builtin_amdgcn_global_load_async_to_lds_b128(
      (__attribute__((address_space(1))) v4i_vs*)gsrc,
      (__attribute__((address_space(3))) v4i_vs*)lds_dst, 0, 0);
#else
  *reinterpret_cast<v8bf*>(lds_dst) = *reinterpret_cast<const v8bf*>(gsrc);
#endif
}

__device__ inline void wait_async_le1() {
#if HAVE_ASYNC_LDS
#if __has_builtin(__builtin_amdgcn_s_wait_asynccnt)
  __builtin_amdgcn_s_wait_asynccnt(1);
#else
  asm volatile("s_wait_asynccnt 0x1" ::: "memory");
#endif
#endif
}

__device__ inline void wait_async_le0() {
#if HAVE_ASYNC_LDS
#if __has_builtin(__builtin_amdgcn_s_wait_asynccnt)
  __builtin_amdgcn_s_wait_asynccnt(0);
#else
  asm volatile("s_wait_asynccnt 0x0" ::: "memory");
#endif
#endif
}

// ---- WMMA fragment loaders (layouts per CDNA5 ISA §7.12.2) ------

// A 16x32 bf16: lane l holds row m=l&15; lanes 0-15 carry K={0..7,16..23},
// lanes 16-31 carry K={8..15,24..31}. Two 16B contiguous loads per lane.
__device__ inline v16bf load_a_frag(const __bf16* __restrict__ A, int lda,
                                    int row0, int k0, int lane) {
  const int m    = lane & 15;
  const int koff = (lane >> 4) * 8;
  const __bf16* base = A + (size_t)(row0 + m) * lda + k0;
  union { v16bf v; v8bf h[2]; } u;
  u.h[0] = *reinterpret_cast<const v8bf*>(base + koff);
  u.h[1] = *reinterpret_cast<const v8bf*>(base + 16 + koff);
  return u.v;
}

// B 32x16 bf16 fragment from an LDS tile stored as tile[n][kpad] (kpad=40):
// lane l holds col n=l&15; lanes 0-15 K={0..7,16..23}, lanes 16-31 K={8..15,24..31}.
#define KPAD 40
__device__ inline v16bf load_b_frag_lds(const __bf16* tile, int n0, int lane) {
  const int n  = n0 + (lane & 15);
  const int kb = (lane >> 4) * 8;
  const __bf16* base = tile + n * KPAD + kb;
  union { v16bf v; v8bf h[2]; } u;
  u.h[0] = *reinterpret_cast<const v8bf*>(base);
  u.h[1] = *reinterpret_cast<const v8bf*>(base + 16);
  return u.v;
}

// ---- Generic bf16 WMMA GEMM: C(MxN) = epi(A(MxK) * B^T),  B row-major NxK --
// 256 threads = 8 wave32; block tile M=128, N=64; each wave: 16x64 strip.
// EPI: 0 = store f32 row-major
//      1 = tanh(acc+bias[col]) -> bf16 row-major into Cp AND transposed into Ct
//      2 = store bf16 transposed only (Cp[col*ldc + row])
template <int EPI, int K>
__global__ __launch_bounds__(256)
void gemm_bf16_wmma(const __bf16* __restrict__ A, size_t sA, int lda,
                    const __bf16* __restrict__ Bm, size_t sB, int ldb,
                    void* __restrict__ Cp, size_t sC, int ldc,
                    __bf16* __restrict__ Ct, size_t sCt, int ldct,
                    const float* __restrict__ bias) {
  constexpr int STEPS = K / 32;
  __shared__ __bf16 Bt[2][64][KPAD];   // double-buffered B tile, bank-padded

  const int tid   = threadIdx.x;
  const int lane  = tid & 31;
  const int wave  = tid >> 5;
  const int batch = blockIdx.z;

  const __bf16* Ab = A  + (size_t)batch * sA;
  const __bf16* Bb = Bm + (size_t)batch * sB;

  const int row0 = blockIdx.y * 128 + wave * 16;
  const int n0   = blockIdx.x * 64;

  // B-tile fill: 256 threads x 16B = 4KB = 64 rows x 32 k-values
  const int frow   = tid >> 2;        // 0..63
  const int fchunk = (tid & 3) * 8;   // 0,8,16,24
  const __bf16* fsrc0 = Bb + (size_t)(n0 + frow) * ldb + fchunk;

  v8f acc[4] = {};

  copy16_g2l(&Bt[0][frow][fchunk], fsrc0);           // prologue fill, k=0

  // steady state: branch-free fill(next) / wait<=1 / barrier / 4x WMMA / barrier
#pragma unroll 1
  for (int it = 0; it < STEPS - 1; ++it) {
    const int cur = it & 1;
    const int k   = it * 32;

    copy16_g2l(&Bt[cur ^ 1][frow][fchunk], fsrc0 + (k + 32));
    wait_async_le1();                  // own fills for `cur` retired (in-order)
    __builtin_prefetch(Ab + (size_t)(row0 + (lane & 15)) * lda + k + 32, 0, 3);
    __syncthreads();                   // all waves' fills for `cur` visible

    v16bf a = load_a_frag(Ab, lda, row0, k, lane);
#pragma unroll
    for (int j = 0; j < 4; ++j) {
      v16bf b = load_b_frag_lds(&Bt[cur][0][0], j * 16, lane);
      acc[j] = __builtin_amdgcn_wmma_f32_16x16x32_bf16(
          /*neg_a=*/false, a, /*neg_b=*/false, b,
          /*c_mod=*/(short)0, acc[j], /*reuse_a=*/false, /*reuse_b=*/false);
    }
    __syncthreads();                   // compute done before buffer overwrite
  }

  // peeled final K-step
  {
    constexpr int it  = STEPS - 1;
    constexpr int cur = it & 1;
    constexpr int k   = it * 32;
    wait_async_le0();
    __syncthreads();

    v16bf a = load_a_frag(Ab, lda, row0, k, lane);
#pragma unroll
    for (int j = 0; j < 4; ++j) {
      v16bf b = load_b_frag_lds(&Bt[cur][0][0], j * 16, lane);
      acc[j] = __builtin_amdgcn_wmma_f32_16x16x32_bf16(
          false, a, false, b, (short)0, acc[j], false, false);
    }
  }

  // C/D layout: VGPR i holds M=i (lanes 0-15) or M=8+i (lanes 16-31); col=lane&15
  const int rbase = row0 + ((lane >> 4) ? 8 : 0);
  const int cln   = lane & 15;
#pragma unroll
  for (int j = 0; j < 4; ++j) {
    const int col = n0 + j * 16 + cln;
#pragma unroll
    for (int i = 0; i < 8; ++i) {
      const int row = rbase + i;
      const float v = acc[j][i];
      if (EPI == 0) {
        reinterpret_cast<float*>(Cp)[(size_t)batch * sC + (size_t)row * ldc + col] = v;
      } else if (EPI == 1) {
        const __bf16 bv = (__bf16)tanhf(v + bias[col]);
        reinterpret_cast<__bf16*>(Cp)[(size_t)batch * sC + (size_t)row * ldc + col] = bv;
        Ct[(size_t)batch * sCt + (size_t)col * ldct + row] = bv;
      } else {
        reinterpret_cast<__bf16*>(Cp)[(size_t)batch * sC + (size_t)col * ldc + row] =
            (__bf16)v;
      }
    }
  }
}

// ---- Prep: x -> Qbf/Dbf/DbfT (bf16), D -> out tail (f32), W -> bf16 --------

__global__ __launch_bounds__(256)
void prep_x(const float* __restrict__ x, __bf16* __restrict__ Qbf,
            __bf16* __restrict__ Dbf, __bf16* __restrict__ DbfT,
            float* __restrict__ out) {
  const size_t idx = (size_t)blockIdx.x * 256 + threadIdx.x;  // B*H*H threads
  const size_t b   = idx / ((size_t)HH * HH);
  const size_t rem = idx % ((size_t)HH * HH);
  const int r = (int)(rem / HH);
  const int h = (int)(rem % HH);
  const float v = x[idx];
  if (r < TT) {
    Qbf[(size_t)b * TT * HH + (size_t)r * HH + h] = (__bf16)v;
  } else {
    const int s = r - TT;
    const __bf16 bv = (__bf16)v;
    Dbf [(size_t)b * TT * HH + (size_t)s * HH + h] = bv;
    DbfT[(size_t)b * TT * HH + (size_t)h * TT + s] = bv;
    out[(size_t)b * TT * OC + (size_t)s * OC + 1536 + h] = v;
  }
}

__global__ __launch_bounds__(256)
void prep_w(const float* __restrict__ W, __bf16* __restrict__ Wbf) {
  const size_t idx = (size_t)blockIdx.x * 256 + threadIdx.x;  // H*H threads
  Wbf[idx] = (__bf16)W[idx];
}

// ---- Softmaxes: one wave32 per row / column with shfl_xor reductions -------

// softmax over s (axis=2) of L[t,:]; write transposed: ADt[s,t] (bf16)
__global__ __launch_bounds__(256)
void softmax_row_T(const float* __restrict__ L, __bf16* __restrict__ ADt) {
  const int lane = threadIdx.x & 31;
  const int wave = threadIdx.x >> 5;
  const int t = blockIdx.x * 8 + wave;
  const int b = blockIdx.y;
  const float* row = L + ((size_t)b * TT + t) * TT;

  float m = -INFINITY;
  for (int s = lane; s < TT; s += 32) m = fmaxf(m, row[s]);
#pragma unroll
  for (int o = 16; o > 0; o >>= 1) m = fmaxf(m, __shfl_xor(m, o, 32));

  float z = 0.f;
  for (int s = lane; s < TT; s += 32) z += __expf(row[s] - m);
#pragma unroll
  for (int o = 16; o > 0; o >>= 1) z += __shfl_xor(z, o, 32);

  const float inv = 1.f / z;
  __bf16* ob = ADt + (size_t)b * TT * TT;
  for (int s = lane; s < TT; s += 32)
    ob[(size_t)s * TT + t] = (__bf16)(__expf(row[s] - m) * inv);
}

// softmax over t (axis=1) of L[:,s]; write row-major: P1[t,s] (bf16)
__global__ __launch_bounds__(256)
void softmax_col(const float* __restrict__ L, __bf16* __restrict__ P1) {
  const int lane = threadIdx.x & 31;
  const int wave = threadIdx.x >> 5;
  const int s = blockIdx.x * 8 + wave;
  const int b = blockIdx.y;
  const float* Lb = L + (size_t)b * TT * TT;

  float m = -INFINITY;
  for (int t = lane; t < TT; t += 32) m = fmaxf(m, Lb[(size_t)t * TT + s]);
#pragma unroll
  for (int o = 16; o > 0; o >>= 1) m = fmaxf(m, __shfl_xor(m, o, 32));

  float z = 0.f;
  for (int t = lane; t < TT; t += 32) z += __expf(Lb[(size_t)t * TT + s] - m);
#pragma unroll
  for (int o = 16; o > 0; o >>= 1) z += __shfl_xor(z, o, 32);

  const float inv = 1.f / z;
  __bf16* ob = P1 + (size_t)b * TT * TT;
  for (int t = lane; t < TT; t += 32)
    ob[(size_t)t * TT + s] = (__bf16)(__expf(Lb[(size_t)t * TT + s] - m) * inv);
}

// ---------------------------------------------------------------------------

extern "C" void kernel_launch(void* const* d_in, const int* in_sizes, int n_in,
                              void* d_out, int out_size, void* d_ws, size_t ws_size,
                              hipStream_t stream) {
  const float* x    = (const float*)d_in[0];
  const float* W    = (const float*)d_in[1];
  const float* bias = (const float*)d_in[2];
  float* out = (float*)d_out;

  // Workspace carving (256B aligned).
  char* ws = (char*)d_ws;
  size_t off = 0;
  auto carve = [&](size_t bytes) -> void* {
    void* p = ws + off;
    off += (bytes + 255) & ~(size_t)255;
    return p;
  };
  const size_t nQ = (size_t)BATCH * TT * HH;   // 384x768 per batch
  const size_t nL = (size_t)BATCH * TT * TT;   // 384x384 per batch

  __bf16* Qbf  = (__bf16*)carve(nQ * 2);
  __bf16* Dbf  = (__bf16*)carve(nQ * 2);
  __bf16* DbfT = (__bf16*)carve(nQ * 2);       // [h][s] per batch
  __bf16* Qa   = (__bf16*)carve(nQ * 2);       // [t][o]
  __bf16* QaT  = (__bf16*)carve(nQ * 2);       // [o][t]
  __bf16* Wbf  = (__bf16*)carve((size_t)HH * HH * 2);
  float*  L    = (float*)carve(nL * 4);
  __bf16* ADt  = (__bf16*)carve(nL * 2);       // [s][t]
  __bf16* P1   = (__bf16*)carve(nL * 2);       // [t][s]
  __bf16* CQ   = (__bf16*)L;                   // reuse: L (f32) == CQ (bf16) bytes, [h][t]

  // 1) precision prep + D pass-through into out[:, :, 1536:]
  prep_x<<<dim3((unsigned)((size_t)BATCH * HH * HH / 256)), 256, 0, stream>>>(
      x, Qbf, Dbf, DbfT, out);
  prep_w<<<dim3((unsigned)((size_t)HH * HH / 256)), 256, 0, stream>>>(W, Wbf);

  // 2) Qa = tanh(Q * W^T + b)  (384x768, K=768); writes Qa[t,o] and QaT[o,t]
  gemm_bf16_wmma<1, HH><<<dim3(HH / 64, TT / 128, BATCH), 256, 0, stream>>>(
      Qbf, (size_t)TT * HH, HH, Wbf, 0, HH,
      Qa, (size_t)TT * HH, HH, QaT, (size_t)TT * HH, TT, bias);

  // 3) L = Qa * D^T  (384x384, K=768), B source = Dbf [s][h]
  gemm_bf16_wmma<0, HH><<<dim3(TT / 64, TT / 128, BATCH), 256, 0, stream>>>(
      Qa, (size_t)TT * HH, HH, Dbf, (size_t)TT * HH, HH,
      L, (size_t)TT * TT, TT, nullptr, 0, 0, nullptr);

  // 4) softmaxes of L
  softmax_row_T<<<dim3(TT / 8, BATCH), 256, 0, stream>>>(L, ADt);  // axis=2 -> ADt[s,t]
  softmax_col<<<dim3(TT / 8, BATCH), 256, 0, stream>>>(L, P1);     // axis=1 -> P1[t,s]

  // 5) CQ[h,t] = sum_s P1[t,s]*D[s,h]  (M=384 t, N=768 h, K=384 s),
  //    B source = DbfT [h][s]; transposed bf16 store
  gemm_bf16_wmma<2, TT><<<dim3(HH / 64, TT / 128, BATCH), 256, 0, stream>>>(
      P1, (size_t)TT * TT, TT, DbfT, (size_t)TT * HH, TT,
      CQ, (size_t)TT * HH, TT, nullptr, 0, 0, nullptr);

  // 6) out[:, :, 0:768] = ADt * QaT^T  (M=384 s, N=768 o, K=384 t), f32 into d_out
  gemm_bf16_wmma<0, TT><<<dim3(HH / 64, TT / 128, BATCH), 256, 0, stream>>>(
      ADt, (size_t)TT * TT, TT, QaT, (size_t)TT * HH, TT,
      out, (size_t)TT * OC, OC, nullptr, 0, 0, nullptr);

  // 7) out[:, :, 768:1536] = ADt * CQ^T  (N cols = h), f32 into d_out
  gemm_bf16_wmma<0, TT><<<dim3(HH / 64, TT / 128, BATCH), 256, 0, stream>>>(
      ADt, (size_t)TT * TT, TT, CQ, (size_t)TT * HH, TT,
      out + 768, (size_t)TT * OC, OC, nullptr, 0, 0, nullptr);
}
